// PGConv2d_18957985644957
// MI455X (gfx1250) — matrix operand order repaired and checked by reading, not compiled, not run on hardware
//
#include <hip/hip_runtime.h>

typedef _Float16 half_t;
typedef __attribute__((ext_vector_type(16))) _Float16 v16h;
typedef __attribute__((ext_vector_type(8)))  _Float16 v8h;
typedef __attribute__((ext_vector_type(8)))  float    v8f;
typedef __attribute__((ext_vector_type(4)))  unsigned u32x4;
typedef __attribute__((ext_vector_type(8)))  int      i32x8;
typedef __attribute__((ext_vector_type(4)))  int      i32x4;

// Problem dims (fixed by the reference)
static constexpr int Nb = 32, Cc = 256, Hh = 56, Ww = 56;
static constexpr int HW = Hh * Ww;            // 3136
static constexpr int M_TOTAL = Nb * HW;       // 100352  (divisible by 64)
static constexpr int K_TOTAL = Cc * 9;        // 2304
static constexpr int KC = 32;                 // K per WMMA chunk
static constexpr int NCHUNK = K_TOTAL / KC;   // 72
static constexpr int X_ELEMS = Nb * Cc * HW;  // 25690112
static constexpr int W_ELEMS = Cc * Cc * 9;   // 589824
static constexpr int LDSROW = 40;             // 32 halfs + 8 pad (80B rows -> conflict-free b128)

// ---------------------------------------------------------------------------
// 1) grid-stride min/max partial reduction
// ---------------------------------------------------------------------------
__global__ void minmax_partial(const float* __restrict__ in, int n,
                               float* __restrict__ pmin, float* __restrict__ pmax) {
  __shared__ float smn[256], smx[256];
  float mn = 3.4e38f, mx = -3.4e38f;
  for (int i = blockIdx.x * 256 + threadIdx.x; i < n; i += gridDim.x * 256) {
    float v = in[i];
    mn = fminf(mn, v);
    mx = fmaxf(mx, v);
  }
  smn[threadIdx.x] = mn; smx[threadIdx.x] = mx;
  for (int s = 128; s > 0; s >>= 1) {
    __syncthreads();
    if (threadIdx.x < s) {
      smn[threadIdx.x] = fminf(smn[threadIdx.x], smn[threadIdx.x + s]);
      smx[threadIdx.x] = fmaxf(smx[threadIdx.x], smx[threadIdx.x + s]);
    }
  }
  __syncthreads();
  if (threadIdx.x == 0) { pmin[blockIdx.x] = smn[0]; pmax[blockIdx.x] = smx[0]; }
}

// ---------------------------------------------------------------------------
// 2) finalize: compute quant params (matches reference fake_quant math)
//    params[0..1]=w8 {scale,zp}  [2..3]=x4  [4..5]=x8
// ---------------------------------------------------------------------------
__global__ void finalize_params(const float* __restrict__ xpmn, const float* __restrict__ xpmx, int nx,
                                const float* __restrict__ wpmn, const float* __restrict__ wpmx, int nw,
                                float* __restrict__ params) {
  __shared__ float smn[256], smx[256];
  const int tid = threadIdx.x;

  float mn = 3.4e38f, mx = -3.4e38f;
  for (int i = tid; i < nx; i += 256) { mn = fminf(mn, xpmn[i]); mx = fmaxf(mx, xpmx[i]); }
  smn[tid] = mn; smx[tid] = mx;
  for (int s = 128; s > 0; s >>= 1) {
    __syncthreads();
    if (tid < s) { smn[tid] = fminf(smn[tid], smn[tid + s]); smx[tid] = fmaxf(smx[tid], smx[tid + s]); }
  }
  __syncthreads();
  const float xmin = smn[0], xmax = smx[0];
  __syncthreads();

  mn = 3.4e38f; mx = -3.4e38f;
  for (int i = tid; i < nw; i += 256) { mn = fminf(mn, wpmn[i]); mx = fmaxf(mx, wpmx[i]); }
  smn[tid] = mn; smx[tid] = mx;
  for (int s = 128; s > 0; s >>= 1) {
    __syncthreads();
    if (tid < s) { smn[tid] = fminf(smn[tid], smn[tid + s]); smx[tid] = fmaxf(smx[tid], smx[tid + s]); }
  }
  __syncthreads();
  const float wmin = smn[0], wmax = smx[0];

  if (tid == 0) {
    const float EPS = 1e-7f;
    float sw = (wmax - wmin) / 255.0f + EPS;  float zw = truncf(-wmin / sw);
    float s4 = (xmax - xmin) / 15.0f  + EPS;  float z4 = truncf(-xmin / s4);
    float s8 = (xmax - xmin) / 255.0f + EPS;  float z8 = truncf(-xmin / s8);
    params[0] = sw; params[1] = zw;
    params[2] = s4; params[3] = z4;
    params[4] = s8; params[5] = z8;
  }
}

__device__ __forceinline__ float fq_apply(float v, float sc, float zp, float qmax) {
  float q = rintf(v / sc + zp);            // round-half-even, like jnp.round
  q = fminf(fmaxf(q, 0.0f), qmax);
  return (q - zp) * sc;
}

// ---------------------------------------------------------------------------
// 3) weight prep: fake-quant(8b) + OIHW -> [co][(r*3+s)*256 + ci] fp16
// ---------------------------------------------------------------------------
__global__ void prep_w(const float* __restrict__ w, const float* __restrict__ params,
                       half_t* __restrict__ wt) {
  const float sc = params[0], zp = params[1];
  for (int i = blockIdx.x * 256 + threadIdx.x; i < W_ELEMS; i += gridDim.x * 256) {
    int s  = i % 3;
    int r  = (i / 3) % 3;
    int ci = (i / 9) % Cc;
    int co = i / (9 * Cc);
    float dq = fq_apply(w[i], sc, zp, 255.0f);
    wt[co * K_TOTAL + (r * 3 + s) * Cc + ci] = (half_t)dq;
  }
}

// ---------------------------------------------------------------------------
// 4) activation prep: NCHW fp32 -> NHWC fp16 msb=fq4(x), lsb=fq8(x)-fq4(x)
// ---------------------------------------------------------------------------
__global__ void prep_x(const float* __restrict__ x, const float* __restrict__ params,
                       half_t* __restrict__ msb, half_t* __restrict__ lsb) {
  const float s4 = params[2], z4 = params[3];
  const float s8 = params[4], z8 = params[5];
  for (int idx = blockIdx.x * 256 + threadIdx.x; idx < X_ELEMS; idx += gridDim.x * 256) {
    int c = idx & (Cc - 1);
    int t = idx >> 8;
    int wq = t % Ww; t /= Ww;
    int hq = t % Hh; t /= Hh;
    int n  = t;
    float v  = x[((n * Cc + c) * Hh + hq) * Ww + wq];
    float m4 = fq_apply(v, s4, z4, 15.0f);
    float m8 = fq_apply(v, s8, z8, 255.0f);
    msb[idx] = (half_t)m4;
    lsb[idx] = (half_t)(m8 - m4);
  }
}

// ---------------------------------------------------------------------------
// TDM: issue a 64(cout) x 32(k) fp16 tile load, global -> padded LDS rows.
// D# built per CDNA5 ISA. pad: 16 DWORDs data + 4 DWORDs
// pad per row => 80-byte LDS rows (LDSROW=40 halfs), matching compute layout.
// 6-arg builtin form (amdgpu-toolchain clang-23): (g0,g1,g2,g3,g4,cpol).
// ---------------------------------------------------------------------------
__device__ __forceinline__ void tdm_load_b_tile(const half_t* gsrc, half_t* ldst) {
  unsigned long long ga = (unsigned long long)(const void*)gsrc;   // global byte addr
  unsigned lds = (unsigned)(unsigned long long)(void*)ldst;        // LDS byte addr (low 32b)
  u32x4 g0 = { 1u,                                   // count=1 (user descriptor)
               lds,                                  // lds_addr
               (unsigned)(ga & 0xffffffffu),         // global_addr[31:0]
               (unsigned)((ga >> 32) & 0x1ffffffu) | (2u << 30) };  // addr[56:32] | type=2
  i32x8 g1 = { (1 << 16) | (1 << 20) | (3 << 22) | (3 << 25),
               //  data_size=2B   pad_en  interval=16dw  amount=4dw ; wg_mask=0
               (int)((unsigned)K_TOTAL << 16),       // tensor_dim0[15:0] (2304)
               (int)(256u << 16),                    // tensor_dim0[31:16]=0 | tensor_dim1=256
               (int)(32u << 16),                     // tensor_dim1[31:16]=0 | tile_dim0=32
               64,                                   // tile_dim1=64 | tile_dim2=0
               K_TOTAL,                              // tensor_dim0_stride[31:0]
               0, 0 };                               // stride hi / dim1_stride = 0
  i32x4 gz4 = { 0, 0, 0, 0 };                        // 2D tensor: groups 2/3 unused
  i32x8 gz8 = { 0, 0, 0, 0, 0, 0, 0, 0 };
  __builtin_amdgcn_tensor_load_to_lds(g0, g1, gz4, gz4, gz8, 0);
}

// ---------------------------------------------------------------------------
// 5+6) dual implicit-GEMM conv: TDM-fed weights (double-buffered) + fused gate
//      block: 64 pixels x 64 cout, 8 wave32, WMMA f32_16x16x32_f16
// ---------------------------------------------------------------------------
__launch_bounds__(256)
__global__ void pgconv_wmma(const half_t* __restrict__ Amsb, const half_t* __restrict__ Alsb,
                            const half_t* __restrict__ Bw, float* __restrict__ out) {
  __shared__ __align__(128) half_t la_m[64 * LDSROW];
  __shared__ __align__(128) half_t la_l[64 * LDSROW];
  __shared__ __align__(128) half_t lb0 [64 * LDSROW];
  __shared__ __align__(128) half_t lb1 [64 * LDSROW];

  const int tid  = threadIdx.x;
  const int mblk = blockIdx.x * 64;
  const int nblk = blockIdx.y * 64;

  // cooperative-loader mapping: thread -> (tile row, 8-half column chunk)
  const int lp   = tid >> 2;          // 0..63
  const int lco  = (tid & 3) * 8;     // 0,8,16,24 (halfs)
  // A-fragment layout: k stored contiguously so each lane's
  // fragment (K{0..15} or K{16..31}) is 16 contiguous halves, matching B.
  const int plco = lco;
  const int m    = mblk + lp;
  const int pn   = m / HW;
  const int phw  = m % HW;
  const int poh  = phw / Ww;
  const int pow_ = phw % Ww;

  // wave/fragment mapping
  const int wave  = tid >> 5;
  const int lane  = tid & 31;
  const int lrow  = lane & 15;
  const int lhi   = lane >> 4;        // 0/1 half-wave
  const int msub  = wave & 3;         // 4 m-subtiles
  const int nsub0 = (wave >> 2) * 2;  // 2 n-subtiles per wave

  const half_t* bbase = Bw + nblk * K_TOTAL;

  v8f acc_m0 = {}, acc_m1 = {}, acc_l0 = {}, acc_l1 = {};

  // prologue: TDM prefetch of weight chunk 0 into buffer 0 (wave 0 issues)
  if (tid < 32) tdm_load_b_tile(bbase, lb0);

  for (int kc = 0; kc < NCHUNK; ++kc) {
    const int rs = kc >> 3;           // 9 taps x 8 channel-chunks
    const int r  = rs / 3;
    const int s  = rs % 3;
    const int c0 = (kc & 7) * 32;

    __syncthreads();                  // prev compute done: A LDS + B[(kc+1)&1] reusable

    // --- A tiles (msb + lsb), zero-padded halo, k-contiguous store
    {
      const int ih = poh + r - 1;
      const int iw = pow_ + s - 1;
      v8h zm = {}, zl = {};
      if ((unsigned)ih < (unsigned)Hh && (unsigned)iw < (unsigned)Ww) {
        const int g = ((pn * Hh + ih) * Ww + iw) * Cc + c0 + lco;
        zm = *(const v8h*)(Amsb + g);
        zl = *(const v8h*)(Alsb + g);
        if (kc + 1 < NCHUNK) {        // next channel chunk -> global_prefetch_b8
          __builtin_prefetch(Amsb + g + KC, 0, 1);
          __builtin_prefetch(Alsb + g + KC, 0, 1);
        }
      }
      *(v8h*)(la_m + lp * LDSROW + plco) = zm;
      *(v8h*)(la_l + lp * LDSROW + plco) = zl;
    }

    // --- B tile for kc+1 via TDM into the other buffer; wait for kc's tile
    if (tid < 32) {
      if (kc + 1 < NCHUNK) {
        tdm_load_b_tile(bbase + (kc + 1) * KC, ((kc + 1) & 1) ? lb1 : lb0);
        __builtin_amdgcn_s_wait_tensorcnt(1);   // tile kc landed, kc+1 in flight
      } else {
        __builtin_amdgcn_s_wait_tensorcnt(0);
      }
    }

    __syncthreads();

    const half_t* lb = (kc & 1) ? lb1 : lb0;

    // --- fragments: 16 contiguous halves per lane, concat two b128 loads
    const int ar = (msub * 16 + lrow) * LDSROW + lhi * 16;
    v8h am_lo = *(const v8h*)(la_m + ar);
    v8h am_hi = *(const v8h*)(la_m + ar + 8);
    v8h al_lo = *(const v8h*)(la_l + ar);
    v8h al_hi = *(const v8h*)(la_l + ar + 8);
    v16h a_m = __builtin_shufflevector(am_lo, am_hi, 0,1,2,3,4,5,6,7,8,9,10,11,12,13,14,15);
    v16h a_l = __builtin_shufflevector(al_lo, al_hi, 0,1,2,3,4,5,6,7,8,9,10,11,12,13,14,15);

    const int br0 = (nsub0 * 16 + lrow) * LDSROW + lhi * 16;
    const int br1 = br0 + 16 * LDSROW;
    v8h b0_lo = *(const v8h*)(lb + br0);
    v8h b0_hi = *(const v8h*)(lb + br0 + 8);
    v8h b1_lo = *(const v8h*)(lb + br1);
    v8h b1_hi = *(const v8h*)(lb + br1 + 8);
    v16h b0 = __builtin_shufflevector(b0_lo, b0_hi, 0,1,2,3,4,5,6,7,8,9,10,11,12,13,14,15);
    v16h b1 = __builtin_shufflevector(b1_lo, b1_hi, 0,1,2,3,4,5,6,7,8,9,10,11,12,13,14,15);

    // --- 4 WMMAs: weight fragments shared between msb/lsb passes
    acc_m0 = __builtin_amdgcn_wmma_f32_16x16x32_f16(false, a_m, false, b0, (short)0, acc_m0, false, false);
    acc_m1 = __builtin_amdgcn_wmma_f32_16x16x32_f16(false, a_m, false, b1, (short)0, acc_m1, false, false);
    acc_l0 = __builtin_amdgcn_wmma_f32_16x16x32_f16(false, a_l, false, b0, (short)0, acc_l0, false, false);
    acc_l1 = __builtin_amdgcn_wmma_f32_16x16x32_f16(false, a_l, false, b1, (short)0, acc_l1, false, false);
  }

  // --- epilogue: out = msb + (sigmoid(msb) > 0.99) * lsb; sigmoid(m)>.99 <=> m>ln(99)
  const float TH_LOGIT = 4.59511985013459f;
  for (int t = 0; t < 2; ++t) {
    v8f am = t ? acc_m1 : acc_m0;
    v8f al = t ? acc_l1 : acc_l0;
    const int co = nblk + (nsub0 + t) * 16 + lrow;
    for (int j = 0; j < 8; ++j) {
      // C/D layout: VGPR j -> M=j (lanes 0-15) / M=j+8 (lanes 16-31), N=lane%16
      const int mrow = mblk + msub * 16 + j + lhi * 8;
      const int n  = mrow / HW;
      const int hw = mrow % HW;
      const float mo = am[j];
      const float o  = mo + (mo > TH_LOGIT ? al[j] : 0.0f);
      out[(n * Cc + co) * HW + hw] = o;
    }
  }
}

// ---------------------------------------------------------------------------
extern "C" void kernel_launch(void* const* d_in, const int* in_sizes, int n_in,
                              void* d_out, int out_size, void* d_ws, size_t ws_size,
                              hipStream_t stream) {
  const float* x = (const float*)d_in[0];   // [32,256,56,56]
  const float* w = (const float*)d_in[1];   // [256,256,3,3]
  float* out = (float*)d_out;

  // workspace layout
  char* ws = (char*)d_ws;
  float*  params = (float*)ws;                          //   64 B
  float*  xpmn   = (float*)(ws + 256);                  // 1024 f
  float*  xpmx   = xpmn + 1024;
  float*  wpmn   = xpmx + 1024;                         //  256 f
  float*  wpmx   = wpmn + 256;
  half_t* wt     = (half_t*)(ws + 16384);               // 2304*256 halfs = 1179648 B
  half_t* msb    = (half_t*)(ws + 16384 + 1179648);     // 25690112 halfs
  half_t* lsb    = msb + X_ELEMS;                       // 25690112 halfs
  (void)in_sizes; (void)n_in; (void)out_size; (void)ws_size;

  minmax_partial<<<1024, 256, 0, stream>>>(x, X_ELEMS, xpmn, xpmx);
  minmax_partial<<<256, 256, 0, stream>>>(w, W_ELEMS, wpmn, wpmx);
  finalize_params<<<1, 256, 0, stream>>>(xpmn, xpmx, 1024, wpmn, wpmx, 256, params);
  prep_w<<<576, 256, 0, stream>>>(w, params, wt);
  prep_x<<<4096, 256, 0, stream>>>(x, params, msb, lsb);

  dim3 grid(M_TOTAL / 64, Cc / 64);  // 1568 x 4
  pgconv_wmma<<<grid, 256, 0, stream>>>(msb, lsb, wt, out);
}
